// Classifier_66357244723416
// MI455X (gfx1250) — 16-bit and 8-bit matrix operand order repaired and checked by reading, compiled, not run on hardware
//
#include <hip/hip_runtime.h>
#include <hip/hip_bf16.h>


typedef __attribute__((ext_vector_type(16))) _Float16 v16h;
typedef __attribute__((ext_vector_type(8)))  float    v8f;
typedef __attribute__((ext_vector_type(8)))  int      v8i;

#define WMMA_F16(a, b, c) \
  __builtin_amdgcn_wmma_f32_16x16x32_f16(false, (a), false, (b), (short)0, (c), false, false)
#define WMMA_FP8(a, b, c) \
  __builtin_amdgcn_wmma_f32_16x16x64_fp8_fp8((a), (b), (short)0, (c), false, false)

#define S_   2048
#define L_   16
#define EC_  64
#define HC_  128
#define EW_  300
#define HW_  512
#define DW_  556
#define DWP_ 576
#define GW_  2048
#define GC_  512

__device__ __forceinline__ float sigmoidf_(float x) { return 1.f / (1.f + __expf(-x)); }

__device__ __forceinline__ unsigned char f32_to_e4m3(float x) {
  union { float f; unsigned u; } v; v.f = x;
  unsigned s = (v.u >> 24) & 0x80u;
  float ax = fabsf(x);
  if (!(ax >= 1.0f / 1024.0f)) return (unsigned char)s;
  if (ax > 448.f) return (unsigned char)(s | 0x7e);
  int e;
  float m = frexpf(ax, &e);
  int exp = e - 1 + 7;
  int mant = (int)((m * 2.f - 1.f) * 8.f + 0.5f);
  if (mant == 8) { mant = 0; ++exp; }
  if (exp <= 0) {
    int dm = (int)(ax * 512.f + 0.5f);
    if (dm > 7) dm = 7;
    return (unsigned char)(s | dm);
  }
  if (exp >= 16) return (unsigned char)(s | 0x7e);
  return (unsigned char)(s | (unsigned)(exp << 3) | (unsigned)mant);
}

__device__ __forceinline__ v16h load_a16(const _Float16* __restrict__ base, int ld,
                                         int m0, int k0) {
  int lane = threadIdx.x & 31;
  int m = lane & 15;
  int hi = lane >> 4;
  const _Float16* p = base + (size_t)(m0 + m) * ld + k0 + hi * 16;
  v16h a;
#pragma unroll
  for (int j = 0; j < 8; ++j) a[j] = p[j];
#pragma unroll
  for (int j = 0; j < 8; ++j) a[8 + j] = p[8 + j];
  return a;
}

__device__ __forceinline__ v16h load_bt16(const _Float16* __restrict__ W, int ldk,
                                          int n0, int k0) {
  int lane = threadIdx.x & 31;
  int n = n0 + (lane & 15);
  int hi = lane >> 4;
  const _Float16* p = W + (size_t)n * ldk + k0 + hi * 16;
  v16h b;
#pragma unroll
  for (int j = 0; j < 16; ++j) b[j] = p[j];
  return b;
}

__device__ __forceinline__ v8i load_a8(const unsigned char* __restrict__ base, int ld,
                                       int m0, int k0) {
  int lane = threadIdx.x & 31;
  const unsigned char* p =
      base + (size_t)(m0 + (lane & 15)) * ld + k0 + (lane >> 4) * 8;
  v8i a;
#pragma unroll
  for (int ch = 0; ch < 4; ++ch) {
    const int* q = (const int*)(p + ch * 16);
    a[ch * 2]     = q[0];
    a[ch * 2 + 1] = q[1];
  }
  return a;
}

__device__ __forceinline__ v8i load_b8vec(const unsigned char* v, int k0) {
  int hi = (threadIdx.x >> 4) & 1;
  const int* q = (const int*)(v + k0 + hi * 8);
  v8i b;
#pragma unroll
  for (int j = 0; j < 8; ++j) b[j] = q[(j >> 1) * 4 + (j & 1)];
  return b;
}

__global__ void cvt_f32_f16(const float* __restrict__ src, _Float16* __restrict__ dst, int n) {
  int i = blockIdx.x * blockDim.x + threadIdx.x;
  if (i < n) dst[i] = (_Float16)src[i];
}

__global__ void cvt_f32_fp8(const float* __restrict__ src, unsigned char* __restrict__ dst, int n) {
  int i = blockIdx.x * blockDim.x + threadIdx.x;
  if (i < n) dst[i] = f32_to_e4m3(src[i]);
}

__global__ void cvt_pad_wih(const float* __restrict__ src, _Float16* __restrict__ dst) {
  int i = blockIdx.x * blockDim.x + threadIdx.x;
  if (i < GW_ * DWP_) {
    int r = i / DWP_, k = i - r * DWP_;
    dst[i] = (k < DW_) ? (_Float16)src[(size_t)r * DW_ + k] : (_Float16)0.f;
  }
}

__global__ void gather_ce(const int* __restrict__ sc, const float* __restrict__ emb,
                          _Float16* __restrict__ ce) {
  int i = blockIdx.x * blockDim.x + threadIdx.x;
  if (i < L_ * S_ * EC_) {
    int e = i & (EC_ - 1);
    int w = (i >> 6) & (S_ - 1);
    int t = i >> 17;
    int ch = sc[w * L_ + t];
    ce[i] = (_Float16)emb[(size_t)ch * EC_ + e];
  }
}

__global__ void pack_xw(const int* __restrict__ sw, const float* __restrict__ wemb,
                        const _Float16* __restrict__ hf, const _Float16* __restrict__ hb,
                        _Float16* __restrict__ xw) {
  int i = blockIdx.x * blockDim.x + threadIdx.x;
  if (i < S_ * DWP_) {
    int w = i / DWP_, k = i - w * DWP_;
    _Float16 v;
    if (k < EW_)            v = (_Float16)wemb[(size_t)sw[w] * EW_ + k];
    else if (k < EW_ + HC_) v = hf[(size_t)w * HC_ + (k - EW_)];
    else if (k < DW_)       v = hb[(size_t)w * HC_ + (k - EW_ - HC_)];
    else                    v = (_Float16)0.f;
    xw[i] = v;
  }
}

__global__ void __launch_bounds__(256)
char_lstm_step(const _Float16* __restrict__ ce_t,
               const _Float16* __restrict__ Wih,
               const _Float16* __restrict__ Whh,
               const float* __restrict__ bih, const float* __restrict__ bhh,
               _Float16* __restrict__ h,
               float* __restrict__ c,
               int first) {
  int wave = threadIdx.x >> 5;
  int lane = threadIdx.x & 31;
  int nlo = lane & 15, hi = lane >> 4;
  int m0 = blockIdx.x * 16;
  int u0 = wave * 16;

  v8f acc[4];
#pragma unroll
  for (int g = 0; g < 4; ++g) {
    int n = g * HC_ + u0 + nlo;
    float b = bih[n] + bhh[n];
#pragma unroll
    for (int r = 0; r < 8; ++r) acc[g][r] = b;
  }
#pragma unroll
  for (int k0 = 0; k0 < EC_; k0 += 32) {
    v16h a = load_a16(ce_t, EC_, m0, k0);
#pragma unroll
    for (int g = 0; g < 4; ++g)
      acc[g] = WMMA_F16(a, load_bt16(Wih, EC_, g * HC_ + u0, k0), acc[g]);
  }
  if (!first) {
#pragma unroll
    for (int k0 = 0; k0 < HC_; k0 += 32) {
      v16h a = load_a16(h, HC_, m0, k0);
#pragma unroll
      for (int g = 0; g < 4; ++g)
        acc[g] = WMMA_F16(a, load_bt16(Whh, HC_, g * HC_ + u0, k0), acc[g]);
    }
  }
  __syncthreads();

  int u = u0 + nlo;
#pragma unroll
  for (int r = 0; r < 8; ++r) {
    int word = m0 + r + 8 * hi;
    size_t idx = (size_t)word * HC_ + u;
    float cp = first ? 0.f : c[idx];
    float iv = sigmoidf_(acc[0][r]);
    float fv = sigmoidf_(acc[1][r]);
    float gv = tanhf(acc[2][r]);
    float ov = sigmoidf_(acc[3][r]);
    float cn = fv * cp + iv * gv;
    c[idx] = cn;
    h[idx] = (_Float16)(ov * tanhf(cn));
  }
}

__global__ void __launch_bounds__(256)
word_xz_gemm(const _Float16* __restrict__ xw,
             const _Float16* __restrict__ Wih,
             const float* __restrict__ bih, const float* __restrict__ bhh,
             float* __restrict__ xz) {
  int wave = threadIdx.x >> 5;
  int lane = threadIdx.x & 31;
  int nlo = lane & 15, hi = lane >> 4;
  int m0 = blockIdx.x * 16;
  int n0 = blockIdx.y * 128 + wave * 16;
  int n = n0 + nlo;
  float b = bih[n] + bhh[n];
  v8f acc;
#pragma unroll
  for (int r = 0; r < 8; ++r) acc[r] = b;
#pragma unroll 2
  for (int k0 = 0; k0 < DWP_; k0 += 32) {
    v16h a = load_a16(xw, DWP_, m0, k0);
    acc = WMMA_F16(a, load_bt16(Wih, DWP_, n0, k0), acc);
  }
#pragma unroll
  for (int r = 0; r < 8; ++r)
    xz[(size_t)(m0 + r + 8 * hi) * GW_ + n] = acc[r];
}

__global__ void __launch_bounds__(1024)
word_lstm(const float* __restrict__ xzf, const float* __restrict__ xzb,
          const unsigned char* __restrict__ Whhf8, const unsigned char* __restrict__ Whhb8,
          float* __restrict__ h_final) {
  __shared__ unsigned char hsh8[HW_];
  __shared__ float xzsh[2][GW_];
  int dir = blockIdx.x;
  const float*         xz  = dir ? xzb : xzf;
  const unsigned char* Whh = dir ? Whhb8 : Whhf8;
  int wave = threadIdx.x >> 5;
  int lane = threadIdx.x & 31;
  int nlo = lane & 15, hi = lane >> 4;
  int u0 = wave * 16;

  unsigned ldsx[2] = { (unsigned)(size_t)&xzsh[0][0], (unsigned)(size_t)&xzsh[1][0] };

  for (int i = threadIdx.x; i < HW_; i += 1024) hsh8[i] = 0;

  {
    int tt0 = dir ? (S_ - 1) : 0;
    const float* g = xz + (size_t)tt0 * GW_ + threadIdx.x * 2;
    asm volatile("global_load_async_to_lds_b64 %0, %1, off"
                 :: "v"(ldsx[0] + threadIdx.x * 8u), "v"(g) : "memory");
  }
  asm volatile("s_wait_asynccnt 0x0" ::: "memory");
  __syncthreads();

  float creg[8], hreg[8];
#pragma unroll
  for (int r = 0; r < 8; ++r) { creg[r] = 0.f; hreg[r] = 0.f; }

  for (int t = 0; t < S_; ++t) {
    int buf = t & 1;
    const float* xzr = &xzsh[buf][0];
    v8f acc[4];
#pragma unroll
    for (int g = 0; g < 4; ++g) {
      int mb = g * HW_ + u0;
#pragma unroll
      for (int r = 0; r < 8; ++r) acc[g][r] = xzr[mb + r + 8 * hi];
    }
    if (t + 1 < S_) {
      int tn = dir ? (S_ - 2 - t) : (t + 1);
      const float* g = xz + (size_t)tn * GW_ + threadIdx.x * 2;
      asm volatile("global_load_async_to_lds_b64 %0, %1, off"
                   :: "v"(ldsx[buf ^ 1] + threadIdx.x * 8u), "v"(g) : "memory");
    }
#pragma unroll 2
    for (int k0 = 0; k0 < HW_; k0 += 64) {
      v8i bv = load_b8vec(hsh8, k0);
#pragma unroll
      for (int g = 0; g < 4; ++g)
        acc[g] = WMMA_FP8(load_a8(Whh, HW_, g * HW_ + u0, k0), bv, acc[g]);
    }
#pragma unroll
    for (int r = 0; r < 8; ++r) {
      float iv = sigmoidf_(acc[0][r]);
      float fv = sigmoidf_(acc[1][r]);
      float gv = tanhf(acc[2][r]);
      float ov = sigmoidf_(acc[3][r]);
      creg[r] = fv * creg[r] + iv * gv;
      hreg[r] = ov * tanhf(creg[r]);
    }
    asm volatile("s_wait_asynccnt 0x0" ::: "memory");
    __syncthreads();
    if (nlo == 0) {
#pragma unroll
      for (int r = 0; r < 8; ++r) hsh8[u0 + r + 8 * hi] = f32_to_e4m3(hreg[r]);
    }
    __syncthreads();
  }
  if (nlo == 0) {
#pragma unroll
    for (int r = 0; r < 8; ++r) h_final[dir * HW_ + u0 + r + 8 * hi] = hreg[r];
  }
}

__global__ void __launch_bounds__(512)
classifier(const float* __restrict__ hf,
           const float* __restrict__ fc1w, const float* __restrict__ fc1b,
           const float* __restrict__ fc2w, const float* __restrict__ fc2b,
           float* __restrict__ out) {
  __shared__ float s1[512];
  __shared__ float lg[20];
  int tid = threadIdx.x;
  float a = fc1b[tid];
  for (int k = 0; k < 1024; ++k) a += hf[k] * fc1w[(size_t)tid * 1024 + k];
  s1[tid] = a > 0.f ? a : 0.f;
  __syncthreads();
  if (tid < 20) {
    float z = fc2b[tid];
    for (int k = 0; k < 512; ++k) z += s1[k] * fc2w[(size_t)tid * 512 + k];
    lg[tid] = z;
  }
  __syncthreads();
  if (tid == 0) {
    float mx = lg[0];
    for (int i = 1; i < 20; ++i) mx = fmaxf(mx, lg[i]);
    float s = 0.f, e[20];
    for (int i = 0; i < 20; ++i) { e[i] = __expf(lg[i] - mx); s += e[i]; }
    for (int i = 0; i < 20; ++i) out[i] = e[i] / s;
  }
}

extern "C" void kernel_launch(void* const* d_in, const int* in_sizes, int n_in,
                              void* d_out, int out_size, void* d_ws, size_t ws_size,
                              hipStream_t stream) {
  const int*   sc      = (const int*)d_in[0];
  const int*   sw      = (const int*)d_in[1];
  const float* cemb    = (const float*)d_in[2];
  const float* cWih_f  = (const float*)d_in[3];
  const float* cWhh_f  = (const float*)d_in[4];
  const float* cbih_f  = (const float*)d_in[5];
  const float* cbhh_f  = (const float*)d_in[6];
  const float* cWih_b  = (const float*)d_in[7];
  const float* cWhh_b  = (const float*)d_in[8];
  const float* cbih_b  = (const float*)d_in[9];
  const float* cbhh_b  = (const float*)d_in[10];
  const float* wemb    = (const float*)d_in[11];
  const float* wWih_f  = (const float*)d_in[12];
  const float* wWhh_f  = (const float*)d_in[13];
  const float* wbih_f  = (const float*)d_in[14];
  const float* wbhh_f  = (const float*)d_in[15];
  const float* wWih_b  = (const float*)d_in[16];
  const float* wWhh_b  = (const float*)d_in[17];
  const float* wbih_b  = (const float*)d_in[18];
  const float* wbhh_b  = (const float*)d_in[19];
  const float* fc1w    = (const float*)d_in[20];
  const float* fc1b    = (const float*)d_in[21];
  const float* fc2w    = (const float*)d_in[22];
  const float* fc2b    = (const float*)d_in[23];
  float* out = (float*)d_out;

  char* ws = (char*)d_ws;
  auto alloc = [&](size_t bytes) -> void* {
    void* p = (void*)ws;
    ws += (bytes + 255) & ~(size_t)255;
    return p;
  };
  _Float16*      ce       = (_Float16*)alloc((size_t)L_ * S_ * EC_ * 2);
  _Float16*      cWihF16f = (_Float16*)alloc((size_t)GC_ * EC_ * 2);
  _Float16*      cWihF16b = (_Float16*)alloc((size_t)GC_ * EC_ * 2);
  _Float16*      cWhhF16f = (_Float16*)alloc((size_t)GC_ * HC_ * 2);
  _Float16*      cWhhF16b = (_Float16*)alloc((size_t)GC_ * HC_ * 2);
  _Float16*      h_cf     = (_Float16*)alloc((size_t)S_ * HC_ * 2);
  _Float16*      h_cb     = (_Float16*)alloc((size_t)S_ * HC_ * 2);
  float*         c_cf     = (float*)alloc((size_t)S_ * HC_ * 4);
  float*         c_cb     = (float*)alloc((size_t)S_ * HC_ * 4);
  _Float16*      xw       = (_Float16*)alloc((size_t)S_ * DWP_ * 2);
  _Float16*      wWihF16f = (_Float16*)alloc((size_t)GW_ * DWP_ * 2);
  _Float16*      wWihF16b = (_Float16*)alloc((size_t)GW_ * DWP_ * 2);
  unsigned char* wWhhF8f  = (unsigned char*)alloc((size_t)GW_ * HW_);
  unsigned char* wWhhF8b  = (unsigned char*)alloc((size_t)GW_ * HW_);
  float*         xzf      = (float*)alloc((size_t)S_ * GW_ * 4);
  float*         xzb      = (float*)alloc((size_t)S_ * GW_ * 4);
  float*         h_fin    = (float*)alloc(1024 * 4);

  const int TB = 256;
  auto blocks = [](long n, int tb) { return (unsigned)((n + tb - 1) / tb); };

  cvt_f32_f16<<<blocks(GC_ * EC_, TB), TB, 0, stream>>>(cWih_f, cWihF16f, GC_ * EC_);
  cvt_f32_f16<<<blocks(GC_ * EC_, TB), TB, 0, stream>>>(cWih_b, cWihF16b, GC_ * EC_);
  cvt_f32_f16<<<blocks(GC_ * HC_, TB), TB, 0, stream>>>(cWhh_f, cWhhF16f, GC_ * HC_);
  cvt_f32_f16<<<blocks(GC_ * HC_, TB), TB, 0, stream>>>(cWhh_b, cWhhF16b, GC_ * HC_);
  cvt_f32_fp8<<<blocks(GW_ * HW_, TB), TB, 0, stream>>>(wWhh_f, wWhhF8f, GW_ * HW_);
  cvt_f32_fp8<<<blocks(GW_ * HW_, TB), TB, 0, stream>>>(wWhh_b, wWhhF8b, GW_ * HW_);
  cvt_pad_wih<<<blocks((long)GW_ * DWP_, TB), TB, 0, stream>>>(wWih_f, wWihF16f);
  cvt_pad_wih<<<blocks((long)GW_ * DWP_, TB), TB, 0, stream>>>(wWih_b, wWihF16b);

  gather_ce<<<blocks((long)L_ * S_ * EC_, TB), TB, 0, stream>>>(sc, cemb, ce);

  for (int s = 0; s < L_; ++s) {
    const _Float16* cef = (const _Float16*)((const char*)ce + (size_t)s * S_ * EC_ * 2);
    const _Float16* ceb = (const _Float16*)((const char*)ce + (size_t)(L_ - 1 - s) * S_ * EC_ * 2);
    char_lstm_step<<<S_ / 16, 256, 0, stream>>>(cef, cWihF16f, cWhhF16f, cbih_f, cbhh_f,
                                                h_cf, c_cf, s == 0);
    char_lstm_step<<<S_ / 16, 256, 0, stream>>>(ceb, cWihF16b, cWhhF16b, cbih_b, cbhh_b,
                                                h_cb, c_cb, s == 0);
  }

  pack_xw<<<blocks((long)S_ * DWP_, TB), TB, 0, stream>>>(sw, wemb, h_cf, h_cb, xw);
  word_xz_gemm<<<dim3(S_ / 16, GW_ / 128), 256, 0, stream>>>(xw, wWihF16f, wbih_f, wbhh_f, xzf);
  word_xz_gemm<<<dim3(S_ / 16, GW_ / 128), 256, 0, stream>>>(xw, wWihF16b, wbih_b, wbhh_b, xzb);

  word_lstm<<<2, 1024, 0, stream>>>(xzf, xzb, wWhhF8f, wWhhF8b, h_fin);

  classifier<<<1, 512, 0, stream>>>(h_fin, fc1w, fc1b, fc2w, fc2b, out);
}
